// Int8Linear_28071906246828
// MI455X (gfx1250) — compile-verified, matrix operand order checked by reading, not run on hardware
//
#include <hip/hip_runtime.h>
#include <cmath>

// ---------------------------------------------------------------------------
// int8-weight linear + exact GELU for MI455X (gfx1250, wave32, WMMA).
//   y[m,n] = gelu( sum_k x[m,k] * (Wq[n,k]*scale[n]) + b[n] )
// M=8192, K=4096, N=11008.  Compute-bound (AI ~1100 FLOP/B), so the kernel is
// built around v_wmma_f32_16x16x32_f16 with f32 accumulation; int8 weights are
// exact in f16, dequant scale folded into the f32 epilogue.
// ---------------------------------------------------------------------------

typedef _Float16 v4h  __attribute__((ext_vector_type(4)));
typedef _Float16 v8h  __attribute__((ext_vector_type(8)));
typedef _Float16 v16h __attribute__((ext_vector_type(16)));
typedef float    v8f  __attribute__((ext_vector_type(8)));

#define M_TOT 8192
#define K_TOT 4096
#define N_TOT 11008

#define BM 128          // block tile M
#define BN 128          // block tile N
#define BK 32           // K chunk staged in LDS (one WMMA K-step)
#define NCHUNK (K_TOT / BK)   // 128

#define LDT 40          // padded LDS row stride in halves (80B: 16B-aligned,
                        // 20 dwords/row, 5 coprime 16 -> conflict-free b128)
#define TILE_H (BM * LDT)     // halves per staged tile (A or B)

// Assemble a 16x32 f16 WMMA fragment from LDS.
// Per ISA A-layout: lane L holds row (L&15); K chunks at (L>>4)*8 and +16.
// B fragment assumed mirror-symmetric (lane holds column n = L&15), which
// matches W stored as (N,K) row-major -> identical addressing as A.
__device__ __forceinline__ v16h frag_ld(const _Float16* p) {
    v8h lo = *(const v8h*)(p);
    v8h hi = *(const v8h*)(p + 16);
    return __builtin_shufflevector(lo, hi,
        0, 1, 2, 3, 4, 5, 6, 7, 8, 9, 10, 11, 12, 13, 14, 15);
}

__device__ __forceinline__ float gelu_exact(float y) {
    return 0.5f * y * (1.0f + erff(y * 0.70710678118654752f));
}

__global__ __launch_bounds__(256)
void int8_linear_gelu_wmma(const float* __restrict__ x,
                           const int*   __restrict__ Wq,
                           const float* __restrict__ scale,
                           const float* __restrict__ bias,
                           float*       __restrict__ out)
{
    // double-buffered A+B tiles: 2 * 2 * 5120 halves = 40 KB
    __shared__ _Float16 smem[2 * 2 * TILE_H];

    const int tid  = threadIdx.x;
    const int lane = tid & 31;
    const int wave = tid >> 5;
    const int wm   = wave & 1;   // 2 waves along M  -> 64 rows each
    const int wn   = wave >> 1;  // 4 waves along N  -> 32 cols each

    const int n0 = blockIdx.x * BN;
    const int m0 = blockIdx.y * BM;

    const int lrow = lane & 15;        // fragment row / col within 16
    const int koff = (lane >> 4) * 8;  // per-half-wave K offset

    // staging decomposition: 128 rows x 8 float4 per row = 1024 vec4,
    // 4 per thread.  cg = tid&7 (fixed column group), rows tid>>3 + i*32.
    const int scg  = tid & 7;
    const int srow = tid >> 3;

    v8f acc[4][2] = {};        // 4 M-subtiles x 2 N-subtiles of 16x16 f32
    float4 areg[4];
    int4   breg[4];

    const size_t kstep = (size_t)BK;

    // ---- prologue: stage chunk 0 --------------------------------------
    {
        const size_t kb = (size_t)scg * 4;
#pragma unroll
        for (int i = 0; i < 4; ++i) {
            const int r = srow + i * 32;
            areg[i] = *(const float4*)(x  + (size_t)(m0 + r) * K_TOT + kb);
            breg[i] = *(const int4*)  (Wq + (size_t)(n0 + r) * K_TOT + kb);
        }
        _Float16* sa = smem;            // buffer 0
        _Float16* sb = smem + TILE_H;
#pragma unroll
        for (int i = 0; i < 4; ++i) {
            const int r = srow + i * 32;
            v4h ha = { (_Float16)areg[i].x, (_Float16)areg[i].y,
                       (_Float16)areg[i].z, (_Float16)areg[i].w };
            v4h hb = { (_Float16)(float)breg[i].x, (_Float16)(float)breg[i].y,
                       (_Float16)(float)breg[i].z, (_Float16)(float)breg[i].w };
            *(v4h*)(sa + r * LDT + scg * 4) = ha;
            *(v4h*)(sb + r * LDT + scg * 4) = hb;
        }
    }
    __syncthreads();

    // ---- main K loop, double buffered ---------------------------------
    for (int kc = 0; kc < NCHUNK; ++kc) {
        const int cur = kc & 1;

        // stage next chunk's global loads (overlaps WMMA below)
        if (kc + 1 < NCHUNK) {
            const size_t kb = (size_t)(kc + 1) * kstep + (size_t)scg * 4;
#pragma unroll
            for (int i = 0; i < 4; ++i) {
                const int r = srow + i * 32;
                areg[i] = *(const float4*)(x  + (size_t)(m0 + r) * K_TOT + kb);
                breg[i] = *(const int4*)  (Wq + (size_t)(n0 + r) * K_TOT + kb);
            }
        }
        // speculative prefetch two chunks ahead (global_prefetch_b8)
        if (kc + 2 < NCHUNK) {
            const size_t kb = (size_t)(kc + 2) * kstep + (size_t)scg * 4;
            __builtin_prefetch(x  + (size_t)(m0 + srow) * K_TOT + kb, 0, 1);
            __builtin_prefetch(Wq + (size_t)(n0 + srow) * K_TOT + kb, 0, 1);
        }

        // ---- compute on current buffer --------------------------------
        {
            const _Float16* sa = smem + cur * (2 * TILE_H)
                               + (wm * 64 + lrow) * LDT + koff;
            const _Float16* sb = smem + cur * (2 * TILE_H) + TILE_H
                               + (wn * 32 + lrow) * LDT + koff;
            v16h af[4], bf[2];
#pragma unroll
            for (int mi = 0; mi < 4; ++mi)
                af[mi] = frag_ld(sa + mi * 16 * LDT);
#pragma unroll
            for (int ni = 0; ni < 2; ++ni)
                bf[ni] = frag_ld(sb + ni * 16 * LDT);
#pragma unroll
            for (int ni = 0; ni < 2; ++ni)
#pragma unroll
                for (int mi = 0; mi < 4; ++mi)
                    acc[mi][ni] = __builtin_amdgcn_wmma_f32_16x16x32_f16(
                        false, af[mi], false, bf[ni],
                        (short)0, acc[mi][ni], false, false);
        }
        __syncthreads();

        // ---- convert + store next chunk into the other buffer ---------
        if (kc + 1 < NCHUNK) {
            _Float16* sa = smem + (cur ^ 1) * (2 * TILE_H);
            _Float16* sb = sa + TILE_H;
#pragma unroll
            for (int i = 0; i < 4; ++i) {
                const int r = srow + i * 32;
                v4h ha = { (_Float16)areg[i].x, (_Float16)areg[i].y,
                           (_Float16)areg[i].z, (_Float16)areg[i].w };
                v4h hb = { (_Float16)(float)breg[i].x,
                           (_Float16)(float)breg[i].y,
                           (_Float16)(float)breg[i].z,
                           (_Float16)(float)breg[i].w };
                *(v4h*)(sa + r * LDT + scg * 4) = ha;
                *(v4h*)(sb + r * LDT + scg * 4) = hb;
            }
        }
        __syncthreads();
    }

    // ---- epilogue: dequant-scale, bias, exact GELU, store f32 ----------
    float scl[2], bia[2];
    int   col[2];
#pragma unroll
    for (int ni = 0; ni < 2; ++ni) {
        col[ni] = n0 + wn * 32 + ni * 16 + lrow;
        scl[ni] = scale[col[ni]];
        bia[ni] = bias[col[ni]];
    }
    const int rbase = m0 + wm * 64 + (lane >> 4) * 8;
#pragma unroll
    for (int mi = 0; mi < 4; ++mi) {
#pragma unroll
        for (int ni = 0; ni < 2; ++ni) {
#pragma unroll
            for (int v = 0; v < 8; ++v) {
                const int row = rbase + mi * 16 + v;   // C layout: M = v + 8*(lane>=16)
                float y = acc[mi][ni][v] * scl[ni] + bia[ni];
                out[(size_t)row * N_TOT + col[ni]] = gelu_exact(y);
            }
        }
    }
}

extern "C" void kernel_launch(void* const* d_in, const int* in_sizes, int n_in,
                              void* d_out, int out_size, void* d_ws, size_t ws_size,
                              hipStream_t stream) {
    (void)in_sizes; (void)n_in; (void)out_size; (void)d_ws; (void)ws_size;
    const float* x     = (const float*)d_in[0];
    const int*   Wq    = (const int*)d_in[1];
    const float* scale = (const float*)d_in[2];
    const float* bias  = (const float*)d_in[3];
    float*       out   = (float*)d_out;

    dim3 grid(N_TOT / BN, M_TOT / BM);   // (86, 64) tiles, exact fit
    int8_linear_gelu_wmma<<<grid, 256, 0, stream>>>(x, Wq, scale, bias, out);
}
